// VectorQuantizer_61632780698110
// MI455X (gfx1250) — matrix operand order repaired and checked by reading, not compiled, not run on hardware
//
#include <hip/hip_runtime.h>
#include <stdint.h>

// ---------------------------------------------------------------------------
// VQ-VAE vector quantizer for MI455X (gfx1250), wave32 + WMMA bf16.
//
// dist(m,n) = ||z_m||^2 + ||e_n||^2 - 2 z_m.e_n ; argmin_n drops ||z_m||^2.
// Score GEMM done as 3 bf16 WMMAs (hi*hi + lo*hi + hi*lo) for ~fp32 accuracy
// at bf16-WMMA rate. Whole working set (<80MB) is L2-resident (192MB), so the
// kernel is compute-bound -> v_wmma_f32_16x16x32_bf16 is the right engine.
// Codebook chunks stream L2 -> LDS through the async-DMA path, double-
// buffered so the copy of chunk c+1 overlaps the WMMAs of chunk c.
// ---------------------------------------------------------------------------

typedef __attribute__((ext_vector_type(16))) __bf16 v16bf;
typedef __attribute__((ext_vector_type(8)))  float  v8f;
typedef __attribute__((ext_vector_type(4)))  int    v4i;

#define AS_GLOBAL __attribute__((address_space(1)))
#define AS_LDS    __attribute__((address_space(3)))

#define D_MODEL 512
#define NUM_EMB 8192
#define N_ROWS  16384
#define NCHUNK  16            // codes per chunk (one WMMA N-tile)
#define NCHUNKS (NUM_EMB / NCHUNK)
#define MTILE   64            // rows per workgroup (4 waves x 16)
#define KCHUNKS (D_MODEL / 32)
#define OUT_ZQ_ELEMS (N_ROWS * D_MODEL)
#define LOSS_SCALE (1.25f / (float)(N_ROWS * D_MODEL))

#if defined(__has_builtin) &&                                        \
    __has_builtin(__builtin_amdgcn_global_load_async_to_lds_b128) && \
    __has_builtin(__builtin_amdgcn_s_wait_asynccnt)
#define HAVE_ASYNC_LDS 1
#else
#define HAVE_ASYNC_LDS 0
#endif

// ---------------------------------------------------------------------------
// Kernel P: split emb into bf16 hi/lo and compute ||e||^2 per code (fp32).
// ---------------------------------------------------------------------------
__global__ __launch_bounds__(256) void vq_prep(const float* __restrict__ emb,
                                               uint16_t* __restrict__ eHiU,
                                               uint16_t* __restrict__ eLoU,
                                               float* __restrict__ enorm) {
  const int row = blockIdx.x;
  const int t = threadIdx.x;
  const float* e = emb + row * D_MODEL;
  __bf16* eHi = (__bf16*)eHiU;
  __bf16* eLo = (__bf16*)eLoU;
  float s = 0.f;
#pragma unroll
  for (int k = 0; k < 2; ++k) {
    const int d = t + k * 256;
    const float x = e[d];
    const __bf16 h = (__bf16)x;
    const float lo = x - (float)h;
    eHi[row * D_MODEL + d] = h;
    eLo[row * D_MODEL + d] = (__bf16)lo;
    s += x * x;
  }
#pragma unroll
  for (int off = 16; off >= 1; off >>= 1) s += __shfl_xor(s, off, 32);
  __shared__ float red[8];
  if ((t & 31) == 0) red[t >> 5] = s;
  __syncthreads();
  if (t == 0) {
    float tot = 0.f;
#pragma unroll
    for (int i = 0; i < 8; ++i) tot += red[i];
    enorm[row] = tot;
  }
}

// ---------------------------------------------------------------------------
// Kernel A: per-row argmin over all 8192 codes via WMMA score tiles.
//   block = 128 threads (4 waves); wave w handles rows blockIdx*64 + w*16.
//   A fragments (z rows, split bf16) live in VGPRs for the whole kernel.
//   Code chunks double-buffered in LDS via async global->LDS copies.
// ---------------------------------------------------------------------------
__global__ __launch_bounds__(128, 1) void vq_dist_argmin(
    const float* __restrict__ z, const uint16_t* __restrict__ eHi,
    const uint16_t* __restrict__ eLo, const float* __restrict__ enorm,
    int* __restrict__ idxOut) {
  __shared__ __align__(32) uint16_t sHi[2][NCHUNK * D_MODEL];
  __shared__ __align__(32) uint16_t sLo[2][NCHUNK * D_MODEL];

  const int tid = threadIdx.x;
  const int lane = tid & 31;
  const int wave = tid >> 5;
  const int rowBase = blockIdx.x * MTILE + wave * 16;
  const int m = lane & 15;          // A: lane (and lane+16) hold row m
  const int khalf = lane >> 4;      // 0: low K-half, 1: high K-half
  const int ksel = khalf * 8;

  // Issue one chunk's async copies (16 x b128 per thread; 16 ASYNCcnt/wave).
  auto issueChunk = [&](int cc, int buf) {
#if HAVE_ASYNC_LDS
    AS_GLOBAL v4i* gh = (AS_GLOBAL v4i*)(eHi + (size_t)cc * (NCHUNK * D_MODEL));
    AS_GLOBAL v4i* gl = (AS_GLOBAL v4i*)(eLo + (size_t)cc * (NCHUNK * D_MODEL));
    AS_LDS v4i* lh = (AS_LDS v4i*)&sHi[buf][0];
    AS_LDS v4i* ll = (AS_LDS v4i*)&sLo[buf][0];
#pragma unroll
    for (int kk = 0; kk < 8; ++kk) {
      const int o = kk * 128 + tid;  // 16-byte units
      __builtin_amdgcn_global_load_async_to_lds_b128(gh + o, lh + o, 0, 0);
      __builtin_amdgcn_global_load_async_to_lds_b128(gl + o, ll + o, 0, 0);
    }
#else
    const uint16_t* gHi = eHi + (size_t)cc * (NCHUNK * D_MODEL);
    const uint16_t* gLo = eLo + (size_t)cc * (NCHUNK * D_MODEL);
#pragma unroll
    for (int kk = 0; kk < 8; ++kk) {
      const int off8 = (kk * 128 + tid) * 8;  // uint16 units, 16B per copy
      *(uint4*)(&sHi[buf][off8]) = *(const uint4*)(gHi + off8);
      *(uint4*)(&sLo[buf][off8]) = *(const uint4*)(gLo + off8);
    }
#endif
  };

  // ---- preload A fragments (16 rows x 512, bf16 hi+lo) into registers ----
  const float* zrow = z + (size_t)(rowBase + m) * D_MODEL;
  v16bf aHi[KCHUNKS], aLo[KCHUNKS];
#pragma unroll
  for (int kc = 0; kc < KCHUNKS; ++kc) {
#pragma unroll
    for (int j = 0; j < 8; ++j) {
      const float x0 = zrow[kc * 32 + 2 * ksel + j];        // K = 2*ksel+j
      const float x1 = zrow[kc * 32 + 2 * ksel + 8 + j];    // K = 2*ksel+8+j
      const __bf16 h0 = (__bf16)x0;
      const __bf16 h1 = (__bf16)x1;
      aHi[kc][j] = h0;
      aHi[kc][8 + j] = h1;
      aLo[kc][j] = (__bf16)(x0 - (float)h0);
      aLo[kc][8 + j] = (__bf16)(x1 - (float)h1);
    }
  }

  float best[8];
  int bidx[8];
#pragma unroll
  for (int i = 0; i < 8; ++i) {
    best[i] = 3.4e38f;
    bidx[i] = 0;
  }

  const int n = lane & 15;  // B/C column owned by this lane

  issueChunk(0, 0);  // prologue: chunk 0 -> buffer 0

  for (int c = 0; c < NCHUNKS; ++c) {
    const int cur = c & 1;
    // Stream chunk c+1 behind the compute of chunk c (wrap keeps the
    // wait-count constant and is deterministic; the extra tail copy is dead).
    issueChunk((c + 1) & (NCHUNKS - 1), cur ^ 1);
#if HAVE_ASYNC_LDS
    // Async loads complete in order: waiting down to the newest 16 (chunk
    // c+1, just issued) guarantees chunk c is fully resident in LDS.
    __builtin_amdgcn_s_wait_asynccnt(16);
#endif
    __syncthreads();  // chunk c visible to all waves; buffers handed over

    const float en = enorm[c * NCHUNK + n];
    const uint16_t* bufHi = &sHi[cur][0];
    const uint16_t* bufLo = &sLo[cur][0];
    const int bbase = n * D_MODEL + khalf * 16;

    v8f acc = {0.f, 0.f, 0.f, 0.f, 0.f, 0.f, 0.f, 0.f};
    // Software-pipeline the B-fragment LDS loads one kc ahead so the
    // waitcnt pass can keep DScnt > 0 under the WMMA group.
    v16bf bH = *(const v16bf*)&bufHi[bbase];
    v16bf bL = *(const v16bf*)&bufLo[bbase];
#pragma unroll
    for (int kc = 0; kc < KCHUNKS; ++kc) {
      v16bf nH, nL;
      if (kc + 1 < KCHUNKS) {
        const int boff = bbase + (kc + 1) * 32;
        nH = *(const v16bf*)&bufHi[boff];
        nL = *(const v16bf*)&bufLo[boff];
      }
      acc = __builtin_amdgcn_wmma_f32_16x16x32_bf16(false, aHi[kc], false, bH,
                                                    (short)0, acc, false, false);
      acc = __builtin_amdgcn_wmma_f32_16x16x32_bf16(false, aLo[kc], false, bH,
                                                    (short)0, acc, false, false);
      acc = __builtin_amdgcn_wmma_f32_16x16x32_bf16(false, aHi[kc], false, bL,
                                                    (short)0, acc, false, false);
      if (kc + 1 < KCHUNKS) {
        bH = nH;
        bL = nL;
      }
    }

    const int code = c * NCHUNK + n;
#pragma unroll
    for (int i = 0; i < 8; ++i) {
      const float d = en - 2.0f * acc[i];  // ||e_n||^2 - 2 z_m.e_n
      if (d < best[i]) {
        best[i] = d;
        bidx[i] = code;
      }
    }
    __syncthreads();  // compute(c) done before chunk c+2 overwrites buf[cur]
  }

  // Min-reduce across the 16 lanes of each half (xor 1,2,4,8 stays in-half).
#pragma unroll
  for (int off = 1; off < 16; off <<= 1) {
#pragma unroll
    for (int i = 0; i < 8; ++i) {
      const float ov = __shfl_xor(best[i], off, 32);
      const int oi = __shfl_xor(bidx[i], off, 32);
      if (ov < best[i] || (ov == best[i] && oi < bidx[i])) {
        best[i] = ov;
        bidx[i] = oi;
      }
    }
  }
  if ((lane & 15) == 0) {
    const int mBase = rowBase + khalf * 8;  // lane0 -> M=0..7, lane16 -> M=8..15
#pragma unroll
    for (int i = 0; i < 8; ++i) idxOut[mBase + i] = bidx[i];
  }
}

// ---------------------------------------------------------------------------
// Loss-slot zero init (runs every launch; deterministic).
// ---------------------------------------------------------------------------
__global__ void vq_zero(float* p) {
  if (threadIdx.x == 0 && blockIdx.x == 0) *p = 0.f;
}

// ---------------------------------------------------------------------------
// Kernel B: gather z_q = emb[idx], write outputs, reduce exact fp32 MSE,
// atomically accumulate scaled loss, emit idx as float.
// ---------------------------------------------------------------------------
__global__ __launch_bounds__(256) void vq_gather(const float* __restrict__ z,
                                                 const float* __restrict__ emb,
                                                 const int* __restrict__ idx,
                                                 float* __restrict__ zqOut,
                                                 float* __restrict__ loss,
                                                 float* __restrict__ idxF) {
  const int row = blockIdx.x;
  const int t = threadIdx.x;
  const int code = idx[row];
  const float* e = emb + (size_t)code * D_MODEL;
  const float* zr = z + (size_t)row * D_MODEL;
  float s = 0.f;
#pragma unroll
  for (int k = 0; k < 2; ++k) {
    const int d = t + k * 256;
    const float q = e[d];
    const float diff = q - zr[d];
    zqOut[(size_t)row * D_MODEL + d] = q;  // z_q_st == z_q numerically
    s += diff * diff;
  }
#pragma unroll
  for (int off = 16; off >= 1; off >>= 1) s += __shfl_xor(s, off, 32);
  __shared__ float red[8];
  if ((t & 31) == 0) red[t >> 5] = s;
  __syncthreads();
  if (t == 0) {
    float tot = 0.f;
#pragma unroll
    for (int i = 0; i < 8; ++i) tot += red[i];
    atomicAdd(loss, tot * LOSS_SCALE);
    idxF[row] = (float)code;
  }
}

// ---------------------------------------------------------------------------
// Launch: prep -> zero loss -> distance/argmin -> gather+loss, all on stream.
// ws layout: e_hi 8MB | e_lo 8MB | enorm 32KB | idx 64KB  (~16.1 MB total).
// out layout: z_q [8388608] | loss [1] | idx-as-float [16384].
// ---------------------------------------------------------------------------
extern "C" void kernel_launch(void* const* d_in, const int* in_sizes, int n_in,
                              void* d_out, int out_size, void* d_ws,
                              size_t ws_size, hipStream_t stream) {
  const float* z = (const float*)d_in[0];
  const float* emb = (const float*)d_in[1];
  float* out = (float*)d_out;
  char* ws = (char*)d_ws;

  uint16_t* eHi = (uint16_t*)(ws);
  uint16_t* eLo = (uint16_t*)(ws + (size_t)NUM_EMB * D_MODEL * 2);
  float* enorm = (float*)(ws + (size_t)NUM_EMB * D_MODEL * 4);
  int* idx = (int*)(ws + (size_t)NUM_EMB * D_MODEL * 4 + NUM_EMB * 4);

  float* zq = out;
  float* loss = out + OUT_ZQ_ELEMS;
  float* idxF = out + OUT_ZQ_ELEMS + 1;

  vq_prep<<<NUM_EMB, 256, 0, stream>>>(emb, eHi, eLo, enorm);
  vq_zero<<<1, 1, 0, stream>>>(loss);
  vq_dist_argmin<<<N_ROWS / MTILE, 128, 0, stream>>>(z, eHi, eLo, enorm, idx);
  vq_gather<<<N_ROWS, 256, 0, stream>>>(z, emb, idx, zq, loss, idxF);
}
